// MaskGate_9594956939721
// MI455X (gfx1250) — matrix operand order repaired and checked by reading, not compiled, not run on hardware
//
#include <hip/hip_runtime.h>
#include <hip/hip_bf16.h>

// ---------------- WMMA types ----------------
typedef __attribute__((ext_vector_type(16))) __bf16        bf16x16;
typedef __attribute__((ext_vector_type(8)))  float         f32x8;
typedef __attribute__((ext_vector_type(4)))  float         f32x4;
typedef __attribute__((ext_vector_type(4)))  unsigned int  u32x4;

// ws layout (ushort offsets): W1sz @ 0 (256*512), W2sz @ 131072 (128*256),
// Wf1sz @ 163840 (64*128). Total 172032 ushorts = 344064 bytes.
#define WS_W2_OFF  131072
#define WS_WF1_OFF 163840
#define WS_BYTES   344064

// fp32 -> bf16 bits, round-to-nearest-even
__device__ __forceinline__ unsigned short f32_to_bf16_bits(float f) {
    union { float f; unsigned int u; } a; a.f = f;
    unsigned int u = a.u + 0x7FFFu + ((a.u >> 16) & 1u);
    return (unsigned short)(u >> 16);
}
__device__ __forceinline__ float bf16_bits_to_f32(unsigned short s) {
    union { unsigned int u; float f; } a; a.u = ((unsigned int)s) << 16;
    return a.f;
}

// D = A*B + C, bf16 inputs, f32 accumulate
__device__ __forceinline__ f32x8 wmma_bf16(bf16x16 a, bf16x16 b, f32x8 c) {
    return __builtin_amdgcn_wmma_f32_16x16x32_bf16(
        false, a, false, b, (short)0, c, false, false);
}

// B-fragment from pre-swizzled bf16 ws: tile = 32 lanes x 16 bf16 (32B/lane)
__device__ __forceinline__ bf16x16 loadB_ws(const unsigned short* wsz,
                                            int tilesK, int nt, int kt, int lane) {
    const unsigned short* p = wsz + ((size_t)(nt * tilesK + kt) * 32 + lane) * 16;
    union { u32x4 q[2]; bf16x16 v; } u;
    u.q[0] = *(const u32x4*)(p);
    u.q[1] = *(const u32x4*)(p + 8);
    return u.v;
}

// B-fragment fallback: 16 consecutive fp32 of one weight row (W is [N,K] row-major)
__device__ __forceinline__ bf16x16 loadB_global(const float* p) {
    f32x4 a0 = *(const f32x4*)(p);
    f32x4 a1 = *(const f32x4*)(p + 4);
    f32x4 a2 = *(const f32x4*)(p + 8);
    f32x4 a3 = *(const f32x4*)(p + 12);
    bf16x16 r;
#pragma unroll
    for (int i = 0; i < 4; ++i) {
        r[i]      = (__bf16)a0[i];
        r[4 + i]  = (__bf16)a1[i];
        r[8 + i]  = (__bf16)a2[i];
        r[12 + i] = (__bf16)a3[i];
    }
    return r;
}

// A-fragment from LDS bf16 row-major [*, ld] (two 16B ds reads)
// Lane layout: lanes 0-15 hold K {0..15}, lanes 16-31 {16..31}
__device__ __forceinline__ bf16x16 loadA_lds(const unsigned short* s, int ld,
                                             int m0, int kb, int lane) {
    const unsigned short* p =
        s + (size_t)(m0 + (lane & 15)) * ld + kb + (lane & 16);
    union { u32x4 q[2]; bf16x16 v; } u;
    u.q[0] = *(const u32x4*)(p);
    u.q[1] = *(const u32x4*)(p + 8);
    return u.v;
}

// bias broadcast into C fragment (C layout: lane%16 selects N column)
__device__ __forceinline__ f32x8 bias_init(const float* b, int nbase, int lane) {
    float v = b[nbase + (lane & 15)];
    f32x8 c;
#pragma unroll
    for (int i = 0; i < 8; ++i) c[i] = v;
    return c;
}

// ReLU + f32->bf16 + scatter accumulator into LDS row-major [*, ld]
__device__ __forceinline__ void store_relu_lds(unsigned short* s, int ld,
                                               int m0, int nbase, int lane,
                                               f32x8 acc) {
    int n  = nbase + (lane & 15);
    int mb = m0 + ((lane & 16) >> 1);
#pragma unroll
    for (int r = 0; r < 8; ++r) {
        float v = acc[r] > 0.0f ? acc[r] : 0.0f;
        s[(size_t)(mb + r) * ld + n] = f32_to_bf16_bits(v);
    }
}

// ---------- pre-kernel: fp32 weights -> bf16 B-fragment-swizzled ws ----------
// One thread per (tile, lane): reads 16 fp32, writes 16 bf16 (32B, coalesced).
__global__ __launch_bounds__(256)
void convert_weights(const float* __restrict__ W1, const float* __restrict__ W2,
                     const float* __restrict__ Wf1, unsigned short* __restrict__ wsz) {
    int t    = blockIdx.x * 256 + threadIdx.x;
    int lane = t & 31;
    int tile = t >> 5;
    const float* W; int Kdim, tilesK, ltile; unsigned short* dst;
    if (tile < 256)      { W = W1;  Kdim = 512; tilesK = 16; dst = wsz;              ltile = tile; }
    else if (tile < 320) { W = W2;  Kdim = 256; tilesK = 8;  dst = wsz + WS_W2_OFF;  ltile = tile - 256; }
    else if (tile < 336) { W = Wf1; Kdim = 128; tilesK = 4;  dst = wsz + WS_WF1_OFF; ltile = tile - 320; }
    else return;
    int nt = ltile / tilesK;
    int kt = ltile % tilesK;
    const float* p = W + (size_t)(nt * 16 + (lane & 15)) * Kdim + kt * 32 + (lane & 16);
    unsigned short* q = dst + ((size_t)ltile * 32 + lane) * 16;
#pragma unroll
    for (int j = 0; j < 16; ++j) q[j] = f32_to_bf16_bits(p[j]);
}

// ---------- main fused MLP kernel: 64 rows/block, 256 threads = 8 wave32 ----------
template <bool USE_WS>
__global__ __launch_bounds__(256)
void mask_gate_mlp(const float* __restrict__ X,
                   const float* __restrict__ W1, const float* __restrict__ b1,
                   const float* __restrict__ W2, const float* __restrict__ b2,
                   const float* __restrict__ Wf1, const float* __restrict__ bf1,
                   const float* __restrict__ Wf2, const float* __restrict__ bf2,
                   const unsigned short* __restrict__ wsz,
                   float* __restrict__ out, int nRows) {
    __shared__ __attribute__((aligned(16))) unsigned short s_x [64 * 512]; // 64 KB
    __shared__ __attribute__((aligned(16))) unsigned short s_h1[64 * 256]; // 32 KB
    __shared__ __attribute__((aligned(16))) unsigned short s_h2[64 * 128]; // 16 KB
    __shared__ __attribute__((aligned(16))) unsigned short s_h3[64 * 64];  //  8 KB

    const int tid     = threadIdx.x;
    const int wave    = tid >> 5;
    const int lane    = tid & 31;
    const int rowBase = blockIdx.x * 64;

    // ---- stage X tile [64,512] fp32 -> bf16 LDS (each element converted once) ----
#pragma unroll 4
    for (int i = 0; i < 32; ++i) {
        int e    = tid + 256 * i;   // float4 index; 128 float4 per row
        int row  = e >> 7;
        int col4 = e & 127;
        int gr   = rowBase + row;
        gr = gr < nRows ? gr : (nRows - 1);
        f32x4 x = *(const f32x4*)(X + (size_t)gr * 512 + col4 * 4);
        unsigned int p0 = (unsigned int)f32_to_bf16_bits(x[0]) |
                          ((unsigned int)f32_to_bf16_bits(x[1]) << 16);
        unsigned int p1 = (unsigned int)f32_to_bf16_bits(x[2]) |
                          ((unsigned int)f32_to_bf16_bits(x[3]) << 16);
        unsigned int* d = (unsigned int*)(s_x + row * 512 + col4 * 4);
        d[0] = p0;
        d[1] = p1;
    }
    __syncthreads();

    // ---------------- Layer 1: [64,512] @ W1^T -> h1 [64,256] ----------------
    {
        const int nt0 = wave * 2;
        f32x8 acc[4][2];
#pragma unroll
        for (int mt = 0; mt < 4; ++mt) {
            acc[mt][0] = bias_init(b1, (nt0 + 0) * 16, lane);
            acc[mt][1] = bias_init(b1, (nt0 + 1) * 16, lane);
        }
        const int bkoff = (lane & 16);
        for (int kt = 0; kt < 16; ++kt) {
            const int kb = kt * 32;
            bf16x16 bf0, bf1v;
            if (USE_WS) {
                bf0  = loadB_ws(wsz, 16, nt0 + 0, kt, lane);
                bf1v = loadB_ws(wsz, 16, nt0 + 1, kt, lane);
            } else {
                bf0  = loadB_global(W1 + (size_t)((nt0 + 0) * 16 + (lane & 15)) * 512 + kb + bkoff);
                bf1v = loadB_global(W1 + (size_t)((nt0 + 1) * 16 + (lane & 15)) * 512 + kb + bkoff);
            }
#pragma unroll
            for (int mt = 0; mt < 4; ++mt) {
                bf16x16 af = loadA_lds(s_x, 512, mt * 16, kb, lane);
                acc[mt][0] = wmma_bf16(af, bf0, acc[mt][0]);
                acc[mt][1] = wmma_bf16(af, bf1v, acc[mt][1]);
            }
        }
#pragma unroll
        for (int mt = 0; mt < 4; ++mt) {
            store_relu_lds(s_h1, 256, mt * 16, (nt0 + 0) * 16, lane, acc[mt][0]);
            store_relu_lds(s_h1, 256, mt * 16, (nt0 + 1) * 16, lane, acc[mt][1]);
        }
    }
    __syncthreads();

    // ---------------- Layer 2: h1 [64,256] @ W2^T -> h2 [64,128] ----------------
    {
        const int nt = wave;
        f32x8 acc[4];
#pragma unroll
        for (int mt = 0; mt < 4; ++mt) acc[mt] = bias_init(b2, nt * 16, lane);
        const int bkoff = (lane & 16);
        for (int kt = 0; kt < 8; ++kt) {
            const int kb = kt * 32;
            bf16x16 bfr;
            if (USE_WS) {
                bfr = loadB_ws(wsz + WS_W2_OFF, 8, nt, kt, lane);
            } else {
                bfr = loadB_global(W2 + (size_t)(nt * 16 + (lane & 15)) * 256 + kb + bkoff);
            }
#pragma unroll
            for (int mt = 0; mt < 4; ++mt) {
                bf16x16 af = loadA_lds(s_h1, 256, mt * 16, kb, lane);
                acc[mt] = wmma_bf16(af, bfr, acc[mt]);
            }
        }
#pragma unroll
        for (int mt = 0; mt < 4; ++mt)
            store_relu_lds(s_h2, 128, mt * 16, nt * 16, lane, acc[mt]);
    }
    __syncthreads();

    // ---------------- Layer 3: h2 [64,128] @ Wf1^T -> h3 [64,64] ----------------
    {
        const int nt  = wave >> 1;
        const int mt0 = (wave & 1) * 2;
        f32x8 acc[2];
        acc[0] = bias_init(bf1, nt * 16, lane);
        acc[1] = acc[0];
        const int bkoff = (lane & 16);
        for (int kt = 0; kt < 4; ++kt) {
            const int kb = kt * 32;
            bf16x16 bfr;
            if (USE_WS) {
                bfr = loadB_ws(wsz + WS_WF1_OFF, 4, nt, kt, lane);
            } else {
                bfr = loadB_global(Wf1 + (size_t)(nt * 16 + (lane & 15)) * 128 + kb + bkoff);
            }
#pragma unroll
            for (int i = 0; i < 2; ++i) {
                bf16x16 af = loadA_lds(s_h2, 128, (mt0 + i) * 16, kb, lane);
                acc[i] = wmma_bf16(af, bfr, acc[i]);
            }
        }
        store_relu_lds(s_h3, 64, (mt0 + 0) * 16, nt * 16, lane, acc[0]);
        store_relu_lds(s_h3, 64, (mt0 + 1) * 16, nt * 16, lane, acc[1]);
    }
    __syncthreads();

    // ---------------- Layer 4 + softmax: h3 [64,64] @ Wf2^T[64,2], VALU ----------------
    if (tid < 64) {
        const int row = rowBase + tid;
        float l0 = bf2[0];
        float l1 = bf2[1];
        const unsigned short* hp = s_h3 + tid * 64;
#pragma unroll 8
        for (int j = 0; j < 64; ++j) {
            float h = bf16_bits_to_f32(hp[j]);
            l0 = fmaf(h, Wf2[j], l0);
            l1 = fmaf(h, Wf2[64 + j], l1);
        }
        float m   = fmaxf(l0, l1);
        float e0  = expf(l0 - m);
        float e1  = expf(l1 - m);
        float inv = 1.0f / (e0 + e1);
        if (row < nRows) {
            out[(size_t)row * 2 + 0] = e0 * inv;
            out[(size_t)row * 2 + 1] = e1 * inv;
        }
    }
}

extern "C" void kernel_launch(void* const* d_in, const int* in_sizes, int n_in,
                              void* d_out, int out_size, void* d_ws, size_t ws_size,
                              hipStream_t stream) {
    const float* X   = (const float*)d_in[0];
    const float* W1  = (const float*)d_in[1];
    const float* b1  = (const float*)d_in[2];
    const float* W2  = (const float*)d_in[3];
    const float* b2  = (const float*)d_in[4];
    const float* Wf1 = (const float*)d_in[5];
    const float* bf1 = (const float*)d_in[6];
    const float* Wf2 = (const float*)d_in[7];
    const float* bf2 = (const float*)d_in[8];
    float* out = (float*)d_out;

    const int nRows  = in_sizes[0] / 512;
    const int blocks = (nRows + 63) / 64;

    if (ws_size >= (size_t)WS_BYTES) {
        unsigned short* wsz = (unsigned short*)d_ws;
        convert_weights<<<42, 256, 0, stream>>>(W1, W2, Wf1, wsz);
        mask_gate_mlp<true><<<blocks, 256, 0, stream>>>(
            X, W1, b1, W2, b2, Wf1, bf1, Wf2, bf2, wsz, out, nRows);
    } else {
        mask_gate_mlp<false><<<blocks, 256, 0, stream>>>(
            X, W1, b1, W2, b2, Wf1, bf1, Wf2, bf2, nullptr, out, nRows);
    }
}
